// PDReconCapsuleRouting_66503273611703
// MI455X (gfx1250) — matrix operand order repaired and checked by reading, not compiled, not run on hardware
//
#include <hip/hip_runtime.h>
#include <hip/hip_bf16.h>
#include <math.h>
#include <stdint.h>

// ---------------------------------------------------------------------------
// Problem constants (from reference)
// ---------------------------------------------------------------------------
#define B_     128
#define NIN_   1152
#define DIN_   8
#define NOUT_  10
#define DOUT_  16
#define DECIN_ 160          // NOUT_*DOUT_
#define PIX_   784          // 1*28*28
#define BO_    (B_*NOUT_)   // 1280 rows into decoder

typedef __attribute__((ext_vector_type(16))) _Float16 v16h;
typedef __attribute__((ext_vector_type(8)))  float    v8f;
typedef __attribute__((ext_vector_type(4)))  float    v4f;

union Frag16 { v16h h; v4f f[2]; };

// ---------------------------------------------------------------------------
// CDNA5 async global->LDS copy (ASYNCcnt-tracked), 16B per lane.
// the vdst VGPR gives the LDS address; flat shared pointers carry the LDS offset
// in their low 32 bits, so truncation gives the right VDST value.
// ---------------------------------------------------------------------------
__device__ __forceinline__ void async_b128(unsigned lds_off, const void* gaddr) {
    asm volatile("global_load_async_to_lds_b128 %0, %1, off"
                 :: "v"(lds_off), "v"(gaddr) : "memory");
}
__device__ __forceinline__ void wait_async4() {   // retire older tile (in-order)
    asm volatile("s_wait_asynccnt 0x4" ::: "memory");
}
__device__ __forceinline__ void wait_async0() {
    asm volatile("s_wait_asynccnt 0x0" ::: "memory");
}

// ---------------------------------------------------------------------------
// Kernel: outcaps = squash( sum_k rc[b,o,k] * hat[b,o,k,:] ), hat recomputed
// inline: hat[b,o,k,d] = sum_i W[o,k,d,i]*x[b,k,i].  grid=1280, block=64.
// ---------------------------------------------------------------------------
__global__ __launch_bounds__(64)
void caps_outcaps_kernel(const float* __restrict__ x,
                         const float* __restrict__ W,
                         const float* __restrict__ rc,
                         int use_rc,
                         float* __restrict__ out)
{
    const int bo = blockIdx.x;
    const int b  = bo / NOUT_;
    const int o  = bo % NOUT_;
    const int t  = threadIdx.x;
    const int d  = t & 15;
    const int part = t >> 4;
    const int KP = NIN_ / 4;

    const float* __restrict__ Wo  = W  + (size_t)o * NIN_ * (DOUT_*DIN_);
    const float* __restrict__ xb  = x  + (size_t)b * NIN_ * DIN_;
    const float* __restrict__ rcb = rc + (size_t)bo * NIN_;

    float s = 0.0f;
    for (int k = part*KP; k < (part+1)*KP; ++k) {
        const float* wk = Wo + (size_t)k*(DOUT_*DIN_) + d*DIN_;
        const float* xk = xb + (size_t)k*DIN_;
        float h = 0.0f;
        #pragma unroll
        for (int i = 0; i < DIN_; ++i) h = fmaf(wk[i], xk[i], h);
        float w = use_rc ? rcb[k] : 1.0f;
        s = fmaf(w, h, s);
    }

    __shared__ float red[64];
    red[t] = s;
    __syncthreads();
    float v = red[d] + red[16+d] + red[32+d] + red[48+d];
    __syncthreads();
    red[d] = v;
    __syncthreads();
    float n2 = 0.0f;
    #pragma unroll
    for (int j = 0; j < 16; ++j) n2 = fmaf(red[j], red[j], n2);
    float n  = sqrtf(n2);
    float sc = n2 / (1.0f + n2) / (n + 1e-8f);
    if (t < 16) out[(size_t)bo*DOUT_ + d] = sc * red[d];
}

// ---------------------------------------------------------------------------
// blog[b,o,k] = sum_i x[b,k,i] * (sum_d oc[b,o,d]*W[o,k,d,i])
// ---------------------------------------------------------------------------
__global__ __launch_bounds__(256)
void caps_blogits_kernel(const float* __restrict__ x,
                         const float* __restrict__ W,
                         const float* __restrict__ oc,
                         float* __restrict__ blog)
{
    const int idx = blockIdx.x * 256 + threadIdx.x;
    if (idx >= B_*NOUT_*NIN_) return;
    const int k  = idx % NIN_;
    const int bo = idx / NIN_;
    const int o  = bo % NOUT_;
    const int b  = bo / NOUT_;

    const float* wk = W + ((size_t)o*NIN_ + k) * (DOUT_*DIN_);
    const float* xk = x + ((size_t)b*NIN_ + k) * DIN_;
    const float* op = oc + (size_t)bo * DOUT_;

    float ti[DIN_];
    #pragma unroll
    for (int i = 0; i < DIN_; ++i) ti[i] = 0.0f;
    #pragma unroll
    for (int dd = 0; dd < DOUT_; ++dd) {
        float od = op[dd];
        #pragma unroll
        for (int i = 0; i < DIN_; ++i) ti[i] = fmaf(od, wk[dd*DIN_+i], ti[i]);
    }
    float r = 0.0f;
    #pragma unroll
    for (int i = 0; i < DIN_; ++i) r = fmaf(ti[i], xk[i], r);
    blog[idx] = r;
}

// ---------------------------------------------------------------------------
// c = scale_coef(blog, num_bg=1) over axis o; one thread per (b,k)
// ---------------------------------------------------------------------------
__global__ __launch_bounds__(256)
void caps_cscale_kernel(const float* __restrict__ blog,
                        float* __restrict__ c)
{
    const int idx = blockIdx.x * 256 + threadIdx.x;
    if (idx >= B_*NIN_) return;
    const int k = idx % NIN_;
    const int b = idx / NIN_;
    const size_t base = (size_t)b * NOUT_ * NIN_ + k;

    float mn = 3.4e38f, mx = -3.4e38f;
    float v[NOUT_-1];
    #pragma unroll
    for (int o = 0; o < NOUT_-1; ++o) {
        v[o] = blog[base + (size_t)o*NIN_];
        mn = fminf(mn, v[o]);
        mx = fmaxf(mx, v[o]);
    }
    float den = fmaxf(mx - mn, 1e-6f);
    #pragma unroll
    for (int o = 0; o < NOUT_-1; ++o)
        c[base + (size_t)o*NIN_] = fmaxf((v[o]-mn)/den, 0.5f);
    c[base + (size_t)(NOUT_-1)*NIN_] = 0.5f;
}

// ---------------------------------------------------------------------------
// masked (f16): row=b*10+o, masked[row,j] = (j/16==o) ? oc[b,o,j%16] : 0
// ---------------------------------------------------------------------------
__global__ __launch_bounds__(256)
void caps_mask_kernel(const float* __restrict__ oc,
                      _Float16* __restrict__ masked)
{
    const int idx = blockIdx.x * 256 + threadIdx.x;
    if (idx >= BO_*DECIN_) return;
    const int j   = idx % DECIN_;
    const int row = idx / DECIN_;
    const int o   = row % NOUT_;
    masked[idx] = ((j >> 4) == o) ? (_Float16)oc[(size_t)row*DOUT_ + (j & 15)]
                                  : (_Float16)0.0f;
}

// ---------------------------------------------------------------------------
// One-time per launch: fp32 weights -> f16, pre-transposed to (N x K) so the
// GEMM's B tile is a pure contiguous copy (no transpose in the hot loop).
// ---------------------------------------------------------------------------
__global__ __launch_bounds__(256)
void caps_w2h_kernel(const float* __restrict__ W, _Float16* __restrict__ Wt,
                     int K, int N)
{
    const int idx = blockIdx.x * 256 + threadIdx.x;
    if (idx >= K*N) return;
    const int k = idx / N;     // coalesced read of W
    const int n = idx % N;
    Wt[(size_t)n*K + k] = (_Float16)W[idx];
}

// ---------------------------------------------------------------------------
// WMMA GEMM, f16 in / fp32 accumulate.  A: MxK f16 row-major.  Bt: NxK f16
// row-major (pre-transposed weights).  64x64 tile / 128 threads (4 waves),
// each wave a 32x32 patch (2x2 WMMA frags).  Tiles staged to LDS with
// double-buffered global_load_async_to_lds_b128 (ASYNCcnt pipelining).
// K multiple of 32, M multiple of 64 (160/512/1024, M=1280: all hold).
// ---------------------------------------------------------------------------
__global__ __launch_bounds__(128)
void caps_gemm_f16_async(const _Float16* __restrict__ A,
                         const _Float16* __restrict__ Bt,
                         const float* __restrict__ bias,
                         float* __restrict__ Cf,          // fp32 out (or null)
                         _Float16* __restrict__ Ch,       // f16 out (or null)
                         int M, int N, int K, int act /*0=relu,1=sigmoid*/)
{
    __shared__ __align__(16) _Float16 As[2][64*40];  // rows stride 40 halves (80B)
    __shared__ __align__(16) _Float16 Bs[2][64*40];
    __shared__ __align__(16) _Float16 dump[8];       // sink for OOB async chunks

    const int tid  = threadIdx.x;
    const int wave = tid >> 5;
    const int lane = tid & 31;
    const int wm = wave >> 1, wn = wave & 1;
    const int hi = lane >> 4, ln = lane & 15;
    const int blockM = blockIdx.y * 64;
    const int blockN = blockIdx.x * 64;

    // Pre-zero OOB B rows once (constant across K steps; async never writes them)
    for (int i = tid; i < 64*40; i += 128) {
        int n = i / 40;
        if (blockN + n >= N) { Bs[0][i] = (_Float16)0.0f; Bs[1][i] = (_Float16)0.0f; }
    }

    v8f zero = {};
    v8f acc[2][2];
    #pragma unroll
    for (int i = 0; i < 2; ++i)
        #pragma unroll
        for (int j = 0; j < 2; ++j) acc[i][j] = zero;

    const unsigned dumpOff = (unsigned)(size_t)&dump[0];

    // Issue one 64x32-half tile pair: exactly 4 async instructions per wave.
    auto issue = [&](int kk, int buf) {
        #pragma unroll
        for (int j = 0; j < 2; ++j) {                 // A: 256 chunks of 16B
            int cid = tid + j*128;
            int r = cid >> 2, ch = cid & 3;
            unsigned lo = (unsigned)(size_t)&As[buf][r*40 + ch*8];
            async_b128(lo, A + (size_t)(blockM + r)*K + kk + ch*8);
        }
        #pragma unroll
        for (int j = 0; j < 2; ++j) {                 // B: 256 chunks of 16B
            int cid = tid + j*128;
            int n = cid >> 2, ch = cid & 3;
            int gn = blockN + n;
            bool ok = (gn < N);
            unsigned lo = ok ? (unsigned)(size_t)&Bs[buf][n*40 + ch*8] : dumpOff;
            int gr = ok ? gn : (N - 1);               // clamped valid address
            async_b128(lo, Bt + (size_t)gr*K + kk + ch*8);
        }
    };

    const int TK = K >> 5;
    issue(0, 0);
    for (int t = 0; t < TK; ++t) {
        const int  cur  = t & 1;
        const bool more = (t + 1) < TK;
        if (more) issue((t+1)*32, cur ^ 1);           // prefetch next tile
        if (more) wait_async4(); else wait_async0();  // retire current tile
        __syncthreads();                              // writes visible to all waves

        Frag16 af[2], bf[2];
        #pragma unroll
        for (int mt = 0; mt < 2; ++mt) {              // A frag: lanes16-31 K 8-15/24-31
            int row = wm*32 + mt*16 + ln;
            int c0  = hi * 8;
            af[mt].f[0] = *(const v4f*)&As[cur][row*40 + c0];
            af[mt].f[1] = *(const v4f*)&As[cur][row*40 + 16 + c0];
        }
        #pragma unroll
        for (int nt = 0; nt < 2; ++nt) {              // B frag: lanes16-31 K 8-15/24-31
            int rn = wn*32 + nt*16 + ln;
            int k0 = hi * 8;
            bf[nt].f[0] = *(const v4f*)&Bs[cur][rn*40 + k0];
            bf[nt].f[1] = *(const v4f*)&Bs[cur][rn*40 + 16 + k0];
        }
        __syncthreads();                              // all reads done before reuse

        #pragma unroll
        for (int mt = 0; mt < 2; ++mt)
            #pragma unroll
            for (int nt = 0; nt < 2; ++nt)
                acc[mt][nt] = __builtin_amdgcn_wmma_f32_16x16x32_f16(
                    false, af[mt].h, false, bf[nt].h,
                    (short)0, acc[mt][nt], false, false);
    }

    // epilogue: element (r,lane) -> M = r + 8*hi, N = ln
    #pragma unroll
    for (int mt = 0; mt < 2; ++mt) {
        #pragma unroll
        for (int nt = 0; nt < 2; ++nt) {
            int gn = blockN + wn*32 + nt*16 + ln;
            if (gn < N) {
                float bv = bias[gn];
                #pragma unroll
                for (int r = 0; r < 8; ++r) {
                    int gm = blockM + wm*32 + mt*16 + hi*8 + r;
                    float v = acc[mt][nt][r] + bv;
                    v = (act == 0) ? fmaxf(v, 0.0f) : 1.0f/(1.0f + expf(-v));
                    if (Ch) Ch[(size_t)gm*N + gn] = (_Float16)v;
                    else    Cf[(size_t)gm*N + gn] = v;
                }
            }
        }
    }
}

// ---------------------------------------------------------------------------
// rscore[row] = -sum_j (x[b,j] - recon[row,j])^2 ; block per row
// ---------------------------------------------------------------------------
__global__ __launch_bounds__(128)
void caps_rscore_kernel(const float* __restrict__ x,
                        const float* __restrict__ recon,
                        float* __restrict__ rscore)
{
    const int row = blockIdx.x;
    const int b   = row / NOUT_;
    const float* xb = x     + (size_t)b  *PIX_;
    const float* rb = recon + (size_t)row*PIX_;
    float s = 0.0f;
    for (int j = threadIdx.x; j < PIX_; j += 128) {
        float d = xb[j] - rb[j];
        s = fmaf(d, d, s);
    }
    __shared__ float red[128];
    red[threadIdx.x] = s;
    __syncthreads();
    for (int off = 64; off > 0; off >>= 1) {
        if (threadIdx.x < off) red[threadIdx.x] += red[threadIdx.x + off];
        __syncthreads();
    }
    if (threadIdx.x == 0) rscore[row] = -red[0];
}

// ---------------------------------------------------------------------------
// r = scale_coef(rscore, num_bg=1) per batch; one thread per (b,o)
// ---------------------------------------------------------------------------
__global__ __launch_bounds__(256)
void caps_rscale_kernel(const float* __restrict__ rscore,
                        float* __restrict__ rbuf)
{
    const int idx = blockIdx.x * 256 + threadIdx.x;
    if (idx >= BO_) return;
    const int o = idx % NOUT_;
    const int b = idx / NOUT_;
    float mn = 3.4e38f, mx = -3.4e38f;
    #pragma unroll
    for (int oo = 0; oo < NOUT_-1; ++oo) {
        float v = rscore[b*NOUT_ + oo];
        mn = fminf(mn, v);
        mx = fmaxf(mx, v);
    }
    float den = fmaxf(mx - mn, 1e-6f);
    float r = (o < NOUT_-1) ? fmaxf((rscore[idx]-mn)/den, 0.5f) : 0.5f;
    rbuf[idx] = r;
}

// ---------------------------------------------------------------------------
// rc[b,o,k] = c[b,o,k] * r[b,o]
// ---------------------------------------------------------------------------
__global__ __launch_bounds__(256)
void caps_rc_kernel(const float* __restrict__ c,
                    const float* __restrict__ rbuf,
                    float* __restrict__ rc)
{
    const int idx = blockIdx.x * 256 + threadIdx.x;
    if (idx >= B_*NOUT_*NIN_) return;
    rc[idx] = c[idx] * rbuf[idx / NIN_];
}

// ---------------------------------------------------------------------------
// Launch
// ---------------------------------------------------------------------------
extern "C" void kernel_launch(void* const* d_in, const int* in_sizes, int n_in,
                              void* d_out, int out_size, void* d_ws, size_t ws_size,
                              hipStream_t stream) {
    (void)in_sizes; (void)n_in; (void)out_size; (void)ws_size;

    const float* incaps = (const float*)d_in[0];
    const float* x_in   = (const float*)d_in[1];
    const float* weight = (const float*)d_in[2];
    const float* W1 = (const float*)d_in[3];
    const float* b1 = (const float*)d_in[4];
    const float* W2 = (const float*)d_in[5];
    const float* b2 = (const float*)d_in[6];
    const float* W3 = (const float*)d_in[7];
    const float* b3 = (const float*)d_in[8];
    float* out = (float*)d_out;

    // workspace carve-up, 64B-aligned chunks (~30 MB total)
    char* base = (char*)d_ws;
    auto alloc = [&](size_t bytes) -> char* {
        char* p = base;
        base += (bytes + 63) & ~(size_t)63;
        return p;
    };
    float*    blog   = (float*)   alloc(sizeof(float)*(size_t)B_*NOUT_*NIN_);
    float*    cbuf   = (float*)   alloc(sizeof(float)*(size_t)B_*NOUT_*NIN_);
    float*    rcbuf  = (float*)   alloc(sizeof(float)*(size_t)B_*NOUT_*NIN_);
    float*    oc     = (float*)   alloc(sizeof(float)*(size_t)BO_*DOUT_);
    float*    recon  = (float*)   alloc(sizeof(float)*(size_t)BO_*PIX_);
    float*    rsc    = (float*)   alloc(sizeof(float)*(size_t)BO_);
    float*    rbuf   = (float*)   alloc(sizeof(float)*(size_t)BO_);
    _Float16* maskh  = (_Float16*)alloc(sizeof(_Float16)*(size_t)BO_*DECIN_);
    _Float16* h1h    = (_Float16*)alloc(sizeof(_Float16)*(size_t)BO_*512);
    _Float16* h2h    = (_Float16*)alloc(sizeof(_Float16)*(size_t)BO_*1024);
    _Float16* W1t    = (_Float16*)alloc(sizeof(_Float16)*(size_t)DECIN_*512);
    _Float16* W2t    = (_Float16*)alloc(sizeof(_Float16)*(size_t)512*1024);
    _Float16* W3t    = (_Float16*)alloc(sizeof(_Float16)*(size_t)1024*PIX_);

    const int NBOK = (B_*NOUT_*NIN_ + 255) / 256;
    const int NBK  = (B_*NIN_ + 255) / 256;
    const int NMSK = (BO_*DECIN_ + 255) / 256;

    // one-time (per launch) weight convert+transpose to f16 (N x K)
    caps_w2h_kernel<<<(DECIN_*512 + 255)/256, 256, 0, stream>>>(W1, W1t, DECIN_, 512);
    caps_w2h_kernel<<<(512*1024 + 255)/256, 256, 0, stream>>>(W2, W2t, 512, 1024);
    caps_w2h_kernel<<<(1024*PIX_ + 255)/256, 256, 0, stream>>>(W3, W3t, 1024, PIX_);

    for (int it = 0; it < 2; ++it) {
        caps_outcaps_kernel<<<BO_, 64, 0, stream>>>(incaps, weight, rcbuf, it, oc);
        caps_blogits_kernel<<<NBOK, 256, 0, stream>>>(incaps, weight, oc, blog);
        caps_cscale_kernel<<<NBK, 256, 0, stream>>>(blog, cbuf);
        caps_mask_kernel<<<NMSK, 256, 0, stream>>>(oc, maskh);

        caps_gemm_f16_async<<<dim3(512/64,  BO_/64), 128, 0, stream>>>(
            maskh, W1t, b1, nullptr, h1h, BO_, 512, DECIN_, 0);
        caps_gemm_f16_async<<<dim3(1024/64, BO_/64), 128, 0, stream>>>(
            h1h, W2t, b2, nullptr, h2h, BO_, 1024, 512, 0);
        caps_gemm_f16_async<<<dim3((PIX_+63)/64, BO_/64), 128, 0, stream>>>(
            h2h, W3t, b3, recon, nullptr, BO_, PIX_, 1024, 1);

        caps_rscore_kernel<<<BO_, 128, 0, stream>>>(x_in, recon, rsc);
        caps_rscale_kernel<<<(BO_+255)/256, 256, 0, stream>>>(rsc, rbuf);
        caps_rc_kernel<<<NBOK, 256, 0, stream>>>(cbuf, rbuf, rcbuf);
    }
    // final squash with last routing coefficients -> (B, NOUT, DOUT) output
    caps_outcaps_kernel<<<BO_, 64, 0, stream>>>(incaps, weight, rcbuf, 1, out);
}
